// Full_GSA_26834955665770
// MI455X (gfx1250) — matrix operand order repaired and checked by reading, not compiled, not run on hardware
//
#include <hip/hip_runtime.h>
#include <hip/hip_bf16.h>

typedef __bf16 bf16_t;
typedef __attribute__((ext_vector_type(16))) __bf16 v16bf;
typedef __attribute__((ext_vector_type(8)))  float  v8f;

#define NH   8
#define KD   128
#define NB   2
#define HW   48
#define NN   (HW*HW)          // 2304
#define CC   1024
#define ROWS (NB*NN)          // 4608

// ---------------- helpers: WMMA fragment loads from LDS ----------------
// A fragment (16x32 bf16): row M = lane%16; K = 16*half + 0..15 contiguous
// (matches load_bfrag k order)
__device__ __forceinline__ v16bf load_afrag(const bf16_t* rowptr, int half) {
    v16bf a;
    *(uint4*)&a       = *(const uint4*)(rowptr + 16*half);
    *((uint4*)&a + 1) = *(const uint4*)(rowptr + 16*half + 8);
    return a;
}
// B fragment (32x16 bf16): col N = lane%16; K = 16*half + 0..15 contiguous
__device__ __forceinline__ v16bf load_bfrag(const bf16_t* colptr_khalf) {
    v16bf b;
    *(uint4*)&b       = *(const uint4*)(colptr_khalf);
    *((uint4*)&b + 1) = *(const uint4*)(colptr_khalf + 8);
    return b;
}
#define WMMA_BF16(A,B,C) __builtin_amdgcn_wmma_f32_16x16x32_bf16(false,(A),false,(B),(short)0,(C),false,false)

// ---------------- CDNA5 async global->LDS copy (ASYNCcnt path) ----------------
// Per-lane 16B copy, bypasses VGPRs. LDS address = low 32 bits of generic ptr
// (ISA 10.2: LDS aperture address truncates to LDS_ADDR.U32).
__device__ __forceinline__ void async_copy16(bf16_t* ldst, const bf16_t* gsrc) {
    unsigned l = (unsigned)(uintptr_t)ldst;
    unsigned long long g = (unsigned long long)(uintptr_t)gsrc;
    asm volatile("global_load_async_to_lds_b128 %0, %1, off"
                 :: "v"(l), "v"(g) : "memory");
}
#define WAIT_ASYNC(imm) asm volatile("s_wait_asynccnt " #imm ::: "memory")

// ---------------- f32 -> bf16 convert ----------------
__global__ void k_cvt_bf16(const float* __restrict__ src, bf16_t* __restrict__ dst, int n) {
    int i = blockIdx.x*256 + threadIdx.x;
    if (i < n) dst[i] = (bf16_t)src[i];
}

// ---------------- fused (attn + lepe) -> bf16 ----------------
__global__ void k_fuse_bf16(const float* __restrict__ a, const float* __restrict__ b,
                            bf16_t* __restrict__ dst, int n) {
    int i = blockIdx.x*256 + threadIdx.x;
    if (i < n) dst[i] = (bf16_t)(a[i] + b[i]);
}

// ---------------- depthwise 5x5 LePE conv ----------------
__global__ void k_lepe(const float* __restrict__ V, const float* __restrict__ w,
                       const float* __restrict__ bc, float* __restrict__ lepe) {
    int idx = blockIdx.x*256 + threadIdx.x;
    if (idx >= NB*HW*HW*CC) return;
    int c = idx & (CC-1);
    int t = idx >> 10;
    int x = t % HW; t /= HW;
    int y = t % HW; int b = t / HW;
    float acc = bc[c];
    #pragma unroll
    for (int dy = 0; dy < 5; ++dy) {
        int yy = y + dy - 2;
        if (yy < 0 || yy >= HW) continue;
        #pragma unroll
        for (int dx = 0; dx < 5; ++dx) {
            int xx = x + dx - 2;
            if (xx < 0 || xx >= HW) continue;
            acc += V[(((size_t)b*HW + yy)*HW + xx)*CC + c] * w[(dy*5 + dx)*CC + c];
        }
    }
    lepe[idx] = acc;
}

// ---------------- fused QKV GEMM + bias + scale + RoPE + layout ----------------
// out = x @ W.T  (W is [cout][cin] row-major, so B tile = W rows = contiguous in cin)
__global__ __launch_bounds__(256) void k_qkv(
    const bf16_t* __restrict__ X,
    const bf16_t* __restrict__ Wqb, const bf16_t* __restrict__ Wkb, const bf16_t* __restrict__ Wvb,
    const float* __restrict__ bq, const float* __restrict__ bk, const float* __restrict__ bv,
    const float* __restrict__ sin_t, const float* __restrict__ cos_t,
    bf16_t* __restrict__ Qh, bf16_t* __restrict__ Kh,
    bf16_t* __restrict__ VhT, float* __restrict__ Vf) {
    __shared__ bf16_t sA[2][128*32];
    __shared__ bf16_t sB[2][128*32];
    const int zi = blockIdx.z;                 // 0=Q 1=K 2=V
    const bf16_t* W = (zi==0) ? Wqb : (zi==1) ? Wkb : Wvb;
    const float* bias = (zi==0) ? bq : (zi==1) ? bk : bv;
    const int rowBase = blockIdx.x * 128;
    const int colBase = blockIdx.y * 128;
    const int tid = threadIdx.x, lane = tid & 31, wave = tid >> 5;
    const int lo = lane & 15, half = lane >> 4;

    // async tile stage: 4 ops/thread (2 for A, 2 for B)
    auto stage = [&](int k0, int bufi) {
        #pragma unroll
        for (int it = 0; it < 2; ++it) {
            int ci = tid + 256*it;
            int r = ci >> 2, kc = (ci & 3)*8;
            async_copy16(&sA[bufi][r*32 + kc], X + (size_t)(rowBase + r)*CC + k0 + kc);
            async_copy16(&sB[bufi][r*32 + kc], W + (size_t)(colBase + r)*CC + k0 + kc);
        }
    };

    v8f acc[8] = {};
    stage(0, 0);
    for (int i = 0; i < 32; ++i) {
        if (i < 31) { stage((i+1)*32, (i+1)&1); WAIT_ASYNC(0x4); }
        else        { WAIT_ASYNC(0x0); }
        __syncthreads();
        const bf16_t* cA = sA[i&1];
        const bf16_t* cB = sB[i&1];
        v16bf afr = load_afrag(cA + (wave*16 + lo)*32, half);
        #pragma unroll
        for (int dt = 0; dt < 8; ++dt) {
            v16bf bfr = load_bfrag(cB + (dt*16 + lo)*32 + 16*half);
            acc[dt] = WMMA_BF16(afr, bfr, acc[dt]);
        }
        __syncthreads();
    }
    const float scaling = 0.08838834764831845f;   // 128^-0.5
    #pragma unroll
    for (int dt = 0; dt < 8; ++dt) {
        int c = colBase + dt*16 + lo;
        int h = c >> 7, d = c & (KD-1);
        float bb = bias[c];
        #pragma unroll
        for (int r = 0; r < 8; ++r) {
            int grow = rowBase + wave*16 + r + 8*half;
            int b = grow / NN, n = grow % NN;
            float val = acc[dt][r] + bb;
            if (zi == 1) val *= scaling;
            if (zi != 2) {
                int y = n / HW, x = n % HW;
                float sn = sin_t[((size_t)(y*HW + x))*KD + d];
                float co = cos_t[((size_t)(y*HW + x))*KD + d];
                float partner = __shfl_xor(val, 1, 32);      // pair element c^1
                val = val*co + ((d & 1) ? -partner : partner)*sn;
                bf16_t* dst = (zi == 0) ? Qh : Kh;
                dst[(((size_t)b*NH + h)*NN + n)*KD + d] = (bf16_t)val;
            } else {
                Vf[(size_t)grow*CC + c] = val;
                VhT[(((size_t)b*NH + h)*KD + d)*NN + n] = (bf16_t)val;
            }
        }
    }
}

// ---------------- flash attention (mask add + online softmax) ----------------
__global__ __launch_bounds__(256) void k_attn(
    const bf16_t* __restrict__ Qh, const bf16_t* __restrict__ Kh,
    const bf16_t* __restrict__ VhT, const float* __restrict__ mask,
    float* __restrict__ Oattn) {
    __shared__ bf16_t sQ[128*128];      // 32 KB, [qrow][d]
    __shared__ bf16_t sK[2][32*128];    // 2x8 KB, [m][d]
    __shared__ bf16_t sV[2][128*32];    // 2x8 KB, [d][m]
    __shared__ bf16_t sP[8*16*32];      //  8 KB, per-wave P staging
    const int b = blockIdx.z, h = blockIdx.y;
    const int q0 = blockIdx.x * 128;
    const int tid = threadIdx.x, lane = tid & 31, wave = tid >> 5;
    const int lo = lane & 15, half = lane >> 4;
    const bf16_t* Qp = Qh + (((size_t)b*NH + h)*NN)*KD;
    const bf16_t* Kp = Kh + (((size_t)b*NH + h)*NN)*KD;
    const bf16_t* Vp = VhT + (((size_t)b*NH + h)*KD)*NN;
    const float*  Mp = mask + (size_t)h*NN*NN;

    // async K/V^T tile stage for one 32-key block: 4 ops/thread
    auto stageKV = [&](int kb, int bufi) {
        #pragma unroll
        for (int it = 0; it < 2; ++it) {
            int ci = tid + 256*it;
            { int r = ci >> 4, kc = (ci & 15)*8;  // K: 32x128
              async_copy16(&sK[bufi][r*128 + kc], Kp + (size_t)(kb + r)*KD + kc); }
            { int dr = ci >> 2, mc = (ci & 3)*8;  // V^T: 128x32
              async_copy16(&sV[bufi][dr*32 + mc], Vp + (size_t)dr*NN + kb + mc); }
        }
    };

    // async Q tile stage (once): 8 ops/thread
    #pragma unroll
    for (int it = 0; it < 8; ++it) {
        int ci = tid + 256*it;
        int r = ci >> 4, kc = (ci & 15)*8;
        async_copy16(&sQ[r*128 + kc], Qp + (size_t)(q0 + r)*KD + kc);
    }

    v8f o[8] = {};
    float mrun[8], lrun[8];
    #pragma unroll
    for (int r = 0; r < 8; ++r) { mrun[r] = -1e30f; lrun[r] = 0.f; }

    stageKV(0, 0);
    const int NKB = NN/32;                       // 72
    for (int i = 0; i < NKB; ++i) {
        const int kb = i*32;
        if (i < NKB-1) { stageKV(kb + 32, (i+1)&1); WAIT_ASYNC(0x4); }
        else           { WAIT_ASYNC(0x0); }
        __syncthreads();
        const bf16_t* cK = sK[i&1];
        const bf16_t* cV = sV[i&1];

        // keep the dominant NT mask stream ahead of consumption
        if (i < NKB-1)
            __builtin_prefetch(Mp + (size_t)(q0 + wave*16 + lo)*NN + kb + 32 + 16*half, 0, 0);

        // S = Q K^T + mask  (two 16x16 column tiles)
        v8f sfr0, sfr1;
        #pragma unroll
        for (int mt = 0; mt < 2; ++mt) {
            v8f c;
            int mcol = kb + mt*16 + lo;
            #pragma unroll
            for (int r = 0; r < 8; ++r) {
                int qq = q0 + wave*16 + r + 8*half;
                c[r] = __builtin_nontemporal_load(Mp + (size_t)qq*NN + mcol);
            }
            #pragma unroll
            for (int ks = 0; ks < KD; ks += 32) {
                v16bf afr = load_afrag(sQ + (wave*16 + lo)*128 + ks, half);
                v16bf bfr = load_bfrag(cK + (mt*16 + lo)*128 + ks + 16*half);
                c = WMMA_BF16(afr, bfr, c);
            }
            if (mt == 0) sfr0 = c; else sfr1 = c;
        }

        // online softmax update
        #pragma unroll
        for (int r = 0; r < 8; ++r) {
            float v = fmaxf(sfr0[r], sfr1[r]);
            #pragma unroll
            for (int o2 = 1; o2 < 16; o2 <<= 1) v = fmaxf(v, __shfl_xor(v, o2, 32));
            float nm = fmaxf(mrun[r], v);
            float alpha = __expf(mrun[r] - nm);
            mrun[r] = nm;
            float p0 = __expf(sfr0[r] - nm);
            float p1 = __expf(sfr1[r] - nm);
            sfr0[r] = p0; sfr1[r] = p1;
            float rs = p0 + p1;
            #pragma unroll
            for (int o2 = 1; o2 < 16; o2 <<= 1) rs += __shfl_xor(rs, o2, 32);
            lrun[r] = lrun[r]*alpha + rs;
            #pragma unroll
            for (int dt = 0; dt < 8; ++dt) o[dt][r] *= alpha;
        }

        // stage P (C-layout -> A-layout) through per-wave LDS
        bf16_t* pw = sP + wave*(16*32);
        #pragma unroll
        for (int r = 0; r < 8; ++r) {
            int rl = r + 8*half;
            pw[rl*32 + lo]      = (bf16_t)sfr0[r];
            pw[rl*32 + 16 + lo] = (bf16_t)sfr1[r];
        }
        asm volatile("s_wait_dscnt 0" ::: "memory");

        v16bf pfr = load_afrag(pw + lo*32, half);
        #pragma unroll
        for (int dt = 0; dt < 8; ++dt) {
            v16bf vfr = load_bfrag(cV + (dt*16 + lo)*32 + 16*half);
            o[dt] = WMMA_BF16(pfr, vfr, o[dt]);
        }
        __syncthreads();
    }

    #pragma unroll
    for (int r = 0; r < 8; ++r) {
        int qq = q0 + wave*16 + r + 8*half;
        float inv = 1.0f / lrun[r];
        #pragma unroll
        for (int dt = 0; dt < 8; ++dt) {
            int d = dt*16 + lo;
            Oattn[((size_t)b*NN + qq)*CC + h*KD + d] = o[dt][r] * inv;
        }
    }
}

// ---------------- output projection GEMM ----------------
__global__ __launch_bounds__(256) void k_oproj(
    const bf16_t* __restrict__ A, const bf16_t* __restrict__ Wob,
    const float* __restrict__ bo, float* __restrict__ out) {
    __shared__ bf16_t sA[2][128*32];
    __shared__ bf16_t sB[2][128*32];
    const int rowBase = blockIdx.x * 128;
    const int colBase = blockIdx.y * 128;
    const int tid = threadIdx.x, lane = tid & 31, wave = tid >> 5;
    const int lo = lane & 15, half = lane >> 4;

    auto stage = [&](int k0, int bufi) {
        #pragma unroll
        for (int it = 0; it < 2; ++it) {
            int ci = tid + 256*it;
            int r = ci >> 2, kc = (ci & 3)*8;
            async_copy16(&sA[bufi][r*32 + kc], A   + (size_t)(rowBase + r)*CC + k0 + kc);
            async_copy16(&sB[bufi][r*32 + kc], Wob + (size_t)(colBase + r)*CC + k0 + kc);
        }
    };

    v8f acc[8] = {};
    stage(0, 0);
    for (int i = 0; i < 32; ++i) {
        if (i < 31) { stage((i+1)*32, (i+1)&1); WAIT_ASYNC(0x4); }
        else        { WAIT_ASYNC(0x0); }
        __syncthreads();
        const bf16_t* cA = sA[i&1];
        const bf16_t* cB = sB[i&1];
        v16bf afr = load_afrag(cA + (wave*16 + lo)*32, half);
        #pragma unroll
        for (int dt = 0; dt < 8; ++dt) {
            v16bf bfr = load_bfrag(cB + (dt*16 + lo)*32 + 16*half);
            acc[dt] = WMMA_BF16(afr, bfr, acc[dt]);
        }
        __syncthreads();
    }
    #pragma unroll
    for (int dt = 0; dt < 8; ++dt) {
        int c = colBase + dt*16 + lo;
        float bb = bo[c];
        #pragma unroll
        for (int r = 0; r < 8; ++r) {
            int grow = rowBase + wave*16 + r + 8*half;
            out[(size_t)grow*CC + c] = acc[dt][r] + bb;
        }
    }
}

extern "C" void kernel_launch(void* const* d_in, const int* in_sizes, int n_in,
                              void* d_out, int out_size, void* d_ws, size_t ws_size,
                              hipStream_t stream) {
    const float* x     = (const float*)d_in[0];
    const float* sin_t = (const float*)d_in[1];
    const float* cos_t = (const float*)d_in[2];
    const float* mask  = (const float*)d_in[3];
    const float* Wq    = (const float*)d_in[4];
    const float* bq    = (const float*)d_in[5];
    const float* Wk    = (const float*)d_in[6];
    const float* bk    = (const float*)d_in[7];
    const float* Wv    = (const float*)d_in[8];
    const float* bv    = (const float*)d_in[9];
    const float* lw    = (const float*)d_in[10];
    const float* lb    = (const float*)d_in[11];
    const float* Wo    = (const float*)d_in[12];
    const float* bo    = (const float*)d_in[13];
    float* out = (float*)d_out;

    char* ws = (char*)d_ws;
    const size_t NXE = (size_t)ROWS*CC;     // 4718592 elems
    const size_t NWE = (size_t)CC*CC;       // 1048576 elems
    size_t off = 0;
    bf16_t* xb  = (bf16_t*)(ws + off); off += NXE*2;
    bf16_t* wqb = (bf16_t*)(ws + off); off += NWE*2;
    bf16_t* wkb = (bf16_t*)(ws + off); off += NWE*2;
    bf16_t* wvb = (bf16_t*)(ws + off); off += NWE*2;
    bf16_t* wob = (bf16_t*)(ws + off); off += NWE*2;
    bf16_t* Qh  = (bf16_t*)(ws + off); off += NXE*2;
    bf16_t* Kh  = (bf16_t*)(ws + off); off += NXE*2;
    bf16_t* VhT = (bf16_t*)(ws + off); off += NXE*2;
    float*  Vf  = (float*)(ws + off);  off += NXE*4;
    float*  lep = (float*)(ws + off);  off += NXE*4;
    float*  att = (float*)(ws + off);  off += NXE*4;
    bf16_t* a2  = (bf16_t*)(ws + off); off += NXE*2;
    (void)ws_size; (void)n_in; (void)in_sizes; (void)out_size;

    // converts
    k_cvt_bf16<<<(int)((NXE + 255)/256), 256, 0, stream>>>(x,  xb,  (int)NXE);
    k_cvt_bf16<<<(int)((NWE + 255)/256), 256, 0, stream>>>(Wq, wqb, (int)NWE);
    k_cvt_bf16<<<(int)((NWE + 255)/256), 256, 0, stream>>>(Wk, wkb, (int)NWE);
    k_cvt_bf16<<<(int)((NWE + 255)/256), 256, 0, stream>>>(Wv, wvb, (int)NWE);
    k_cvt_bf16<<<(int)((NWE + 255)/256), 256, 0, stream>>>(Wo, wob, (int)NWE);

    // fused QKV + RoPE
    k_qkv<<<dim3(ROWS/128, CC/128, 3), 256, 0, stream>>>(
        xb, wqb, wkb, wvb, bq, bk, bv, sin_t, cos_t, Qh, Kh, VhT, Vf);

    // LePE depthwise conv
    k_lepe<<<(int)((NXE + 255)/256), 256, 0, stream>>>(Vf, lw, lb, lep);

    // flash attention
    k_attn<<<dim3(NN/128, NH, NB), 256, 0, stream>>>(Qh, Kh, VhT, mask, att);

    // fuse + output projection
    k_fuse_bf16<<<(int)((NXE + 255)/256), 256, 0, stream>>>(att, lep, a2, (int)NXE);
    k_oproj<<<dim3(ROWS/128, CC/128), 256, 0, stream>>>(a2, wob, bo, out);
}
